// SelfAttention_49667001811806
// MI455X (gfx1250) — matrix operand order repaired and checked by reading, not compiled, not run on hardware
//
#include <hip/hip_runtime.h>

#define Bsz  2
#define Sq   2048
#define Dm   2048
#define Hh   16
#define HDm  128
#define HALF 64

typedef __attribute__((ext_vector_type(16))) __bf16        v16bf;
typedef __attribute__((ext_vector_type(8)))  float         v8f;
typedef __attribute__((ext_vector_type(8)))  unsigned int  v8u;

__device__ __forceinline__ unsigned short f2bf(float f) {
  unsigned int u = __builtin_bit_cast(unsigned int, f);
  u += 0x7fffu + ((u >> 16) & 1u);          // round-to-nearest-even
  return (unsigned short)(u >> 16);
}
__device__ __forceinline__ unsigned int pack2bf(float lo, float hi) {
  return (unsigned int)f2bf(lo) | ((unsigned int)f2bf(hi) << 16);
}

__device__ __forceinline__ v8f wmma_bf16(v16bf a, v16bf b, v8f c) {
  // (neg_a, A, neg_b, B, c_mod, C, reuse_a, reuse_b)
  return __builtin_amdgcn_wmma_f32_16x16x32_bf16(false, a, false, b, (short)0, c, false, false);
}

// CDNA5 async copy: 16B global -> LDS, bypasses VGPRs, tracked with ASYNCcnt.
__device__ __forceinline__ void async_copy_b128(void* lds_ptr, const void* gptr) {
  unsigned lds_off = (unsigned)(unsigned long long)lds_ptr;  // LDS aperture: offset in addr[31:0]
  asm volatile("global_load_async_to_lds_b128 %0, %1, off"
               :
               : "v"(lds_off), "v"(gptr)
               : "memory");
}
__device__ __forceinline__ void wait_async0() {
  asm volatile("s_wait_asynccnt 0x0" ::: "memory");
}

// A-fragment (16x32 bf16). base = &row[kbase], kbase = kk*32 + (lane>>4)*16.
// VGPR 0..3 hold K = kbase+{0..7}, VGPR 4..7 hold K = kbase+8+{0..7}.
__device__ __forceinline__ v16bf frag_a(const unsigned short* base) {
  v8u u;
#pragma unroll
  for (int i = 0; i < 4; ++i) {
    u[i]     = *reinterpret_cast<const unsigned int*>(base + 2 * i);
    u[i + 4] = *reinterpret_cast<const unsigned int*>(base + 8 + 2 * i);
  }
  return __builtin_bit_cast(v16bf, u);
}
// B-fragment (32x16 bf16). base = &col[kbase], kbase = kk*32 + (lane>>4)*16.
// VGPR v holds K = kbase + 2v, 2v+1 (16 consecutive K per lane half).
__device__ __forceinline__ v16bf frag_b(const unsigned short* base) {
  v8u u;
#pragma unroll
  for (int i = 0; i < 8; ++i)
    u[i] = *reinterpret_cast<const unsigned int*>(base + 2 * i);
  return __builtin_bit_cast(v16bf, u);
}

// ---------------------------------------------------------------- fp32 -> bf16
__global__ __launch_bounds__(256) void cvt4_kernel(const float* __restrict__ in,
                                                   unsigned short* __restrict__ out,
                                                   int n4) {
  int i = blockIdx.x * 256 + threadIdx.x;
  if (i < n4) {
    float4 f = reinterpret_cast<const float4*>(in)[i];
    uint2 r;
    r.x = pack2bf(f.x, f.y);
    r.y = pack2bf(f.z, f.w);
    reinterpret_cast<uint2*>(out)[i] = r;
  }
}

// ------------------------------------------------- C = A(MxK) * W(NxK)^T + bias
__global__ __launch_bounds__(256)
void gemm_bf16_wmma(const unsigned short* __restrict__ A,
                    const unsigned short* __restrict__ W,
                    const float* __restrict__ bias,
                    void* __restrict__ Cout,
                    int M, int N, int K, int store_bf16) {
  __shared__ __align__(16) unsigned short ldsA[128 * 72];
  __shared__ __align__(16) unsigned short ldsB[128 * 72];
  const int t = threadIdx.x, lane = t & 31, wid = t >> 5;
  const int bm = blockIdx.x * 128, bn = blockIdx.y * 128;
  const int wm = (wid & 3) * 32, wn = (wid >> 2) * 64;

  const v8f vzero = {0.f, 0.f, 0.f, 0.f, 0.f, 0.f, 0.f, 0.f};
  v8f acc[2][4];
#pragma unroll
  for (int mt = 0; mt < 2; ++mt)
#pragma unroll
    for (int nt = 0; nt < 4; ++nt) acc[mt][nt] = vzero;

  for (int k0 = 0; k0 < K; k0 += 64) {
    // 128 rows x 64 cols bf16 per matrix; all 8 copies in flight on ASYNCcnt
#pragma unroll
    for (int i = 0; i < 4; ++i) {
      int seg = t + i * 256, row = seg >> 3, c = (seg & 7) * 8;
      async_copy_b128(ldsA + row * 72 + c, A + (size_t)(bm + row) * K + k0 + c);
      async_copy_b128(ldsB + row * 72 + c, W + (size_t)(bn + row) * K + k0 + c);
    }
    if (k0 + 64 < K) {                   // warm WGP$ for the next K-slice
#pragma unroll
      for (int i = 0; i < 4; ++i) {
        int seg = t + i * 256, row = seg >> 3, c = (seg & 7) * 8;
        __builtin_prefetch(A + (size_t)(bm + row) * K + k0 + 64 + c, 0, 1);
        __builtin_prefetch(W + (size_t)(bn + row) * K + k0 + 64 + c, 0, 1);
      }
    }
    wait_async0();
    __syncthreads();
#pragma unroll
    for (int kk = 0; kk < 2; ++kk) {
      v16bf af[2];
#pragma unroll
      for (int mt = 0; mt < 2; ++mt)
        af[mt] = frag_a(ldsA + (wm + mt * 16 + (lane & 15)) * 72 + kk * 32 + (lane >> 4) * 16);
#pragma unroll
      for (int nt = 0; nt < 4; ++nt) {
        v16bf bf = frag_b(ldsB + (wn + nt * 16 + (lane & 15)) * 72 + kk * 32 + (lane >> 4) * 16);
        acc[0][nt] = wmma_bf16(af[0], bf, acc[0][nt]);
        acc[1][nt] = wmma_bf16(af[1], bf, acc[1][nt]);
      }
    }
    __syncthreads();
  }

  const int rh = (lane >> 4) * 8;
#pragma unroll
  for (int nt = 0; nt < 4; ++nt) {
    const int col = bn + wn + nt * 16 + (lane & 15);
    const float bb = bias[col];
#pragma unroll
    for (int mt = 0; mt < 2; ++mt) {
#pragma unroll
      for (int r = 0; r < 8; ++r) {
        int row = bm + wm + mt * 16 + rh + r;
        float v = acc[mt][nt][r] + bb;
        if (store_bf16)
          ((unsigned short*)Cout)[(size_t)row * N + col] = f2bf(v);
        else
          ((float*)Cout)[(size_t)row * N + col] = v;
      }
    }
  }
}

// ----------------------------------------- RMSNorm (full D) + interleaved RoPE
__global__ __launch_bounds__(256)
void rmsnorm_rope_kernel(const float* __restrict__ X, const float* __restrict__ w,
                         const float* __restrict__ cosb, const float* __restrict__ sinb,
                         unsigned short* __restrict__ out) {
  const int row = blockIdx.x;                 // b*Sq + s
  const int s = row & (Sq - 1);
  const float* x = X + (size_t)row * Dm;
  const int t = threadIdx.x;

  float vx[8];
  float ss = 0.f;
#pragma unroll
  for (int i = 0; i < 4; ++i) {               // 4 pairs per thread, 1024 pairs
    int p = t + i * 256;
    float2 ab = reinterpret_cast<const float2*>(x)[p];
    vx[2 * i] = ab.x;
    vx[2 * i + 1] = ab.y;
    ss += ab.x * ab.x + ab.y * ab.y;
  }
#pragma unroll
  for (int off = 16; off >= 1; off >>= 1) ss += __shfl_xor(ss, off, 32);
  __shared__ float red[8];
  if ((t & 31) == 0) red[t >> 5] = ss;
  __syncthreads();
  float tot = red[0] + red[1] + red[2] + red[3] + red[4] + red[5] + red[6] + red[7];
  const float inv = rsqrtf(tot * (1.f / (float)Dm) + 1e-6f);

#pragma unroll
  for (int i = 0; i < 4; ++i) {
    int p = t + i * 256;
    int j = p & (HALF - 1);
    float c = cosb[s * HALF + j], sn = sinb[s * HALF + j];
    float y1 = vx[2 * i] * inv * w[2 * p];
    float y2 = vx[2 * i + 1] * inv * w[2 * p + 1];
    float o1 = y1 * c - y2 * sn;
    float o2 = y1 * sn + y2 * c;
    *reinterpret_cast<unsigned int*>(out + (size_t)row * Dm + 2 * p) = pack2bf(o1, o2);
  }
}

// ---------------------------------------- flash attention, one (b,h,128-row) WG
__global__ __launch_bounds__(256)
void attn_wmma(const unsigned short* __restrict__ Q,
               const unsigned short* __restrict__ Kd,
               const unsigned short* __restrict__ V,
               unsigned short* __restrict__ O) {
  __shared__ __align__(16) unsigned short ldsK[64 * 136];    // 64 keys x 128 dims
  __shared__ __align__(16) unsigned short ldsVt[128 * 72];   // transposed: [dim][key]
  __shared__ __align__(16) unsigned short ldsP[8][16 * 72];  // per-wave P staging

  const int t = threadIdx.x, lane = t & 31, wid = t >> 5;
  const int mb = blockIdx.x & 15;            // Sq/128 = 16
  const int bh = blockIdx.x >> 4;
  const int b = bh >> 4, h = bh & 15;
  const int m0 = mb * 128 + wid * 16;        // this wave's 16 query rows
  const int rh = (lane >> 4) * 8;
  const size_t headoff = (size_t)h * HDm;

  // Q fragments for all 4 k-steps (HD=128) kept in registers
  v16bf qf[4];
  {
    const unsigned short* qp =
        Q + ((size_t)(b * Sq) + m0 + (lane & 15)) * Dm + headoff;
#pragma unroll
    for (int ks = 0; ks < 4; ++ks)
      qf[ks] = frag_a(qp + ks * 32 + (lane >> 4) * 16);
  }

  const v8f vzero = {0.f, 0.f, 0.f, 0.f, 0.f, 0.f, 0.f, 0.f};
  v8f o[8];
#pragma unroll
  for (int dt = 0; dt < 8; ++dt) o[dt] = vzero;
  float mrun[8], lrun[8];
#pragma unroll
  for (int r = 0; r < 8; ++r) { mrun[r] = -1e30f; lrun[r] = 0.f; }
  const float scale = 0.0883883476483184f;   // 1/sqrt(128)

  for (int j0 = 0; j0 < Sq; j0 += 64) {
    // K block (64 keys x dims 0..63 bf16) via async copy, all segments in flight
#pragma unroll
    for (int i = 0; i < 2; ++i) {
      int seg = t + i * 256, row = seg >> 3, c = (seg & 7) * 8;
      async_copy_b128(ldsK + row * 136 + c,
                      Kd + (size_t)(b * Sq + j0 + row) * Dm + headoff + c);
    }
    // V block transposed into [dim][key]; gather all dwords first, then store
    unsigned int uv[16];
#pragma unroll
    for (int i = 0; i < 16; ++i) {           // 4096 pairs total
      int p = t + i * 256;
      int j = p >> 6, d = (p & 63) * 2;
      uv[i] = *reinterpret_cast<const unsigned int*>(
          V + (size_t)(b * Sq + j0 + j) * Dm + headoff + d);
    }
#pragma unroll
    for (int i = 0; i < 16; ++i) {
      int p = t + i * 256;
      int j = p >> 6, d = (p & 63) * 2;
      ldsVt[d * 72 + j] = (unsigned short)uv[i];
      ldsVt[(d + 1) * 72 + j] = (unsigned short)(uv[i] >> 16);
    }
    wait_async0();
    __syncthreads();

    // scores: 16 rows x 64 keys per wave
    v8f s[4];
#pragma unroll
    for (int nt = 0; nt < 4; ++nt) {
      v8f sa = vzero;
#pragma unroll
      for (int ks = 0; ks < 4; ++ks) {
        v16bf bf = frag_b(ldsK + (nt * 16 + (lane & 15)) * 136 + ks * 32 + (lane >> 4) * 16);
        sa = wmma_bf16(qf[ks], bf, sa);
      }
      s[nt] = sa;
    }

    // online softmax (row r of this tile lives in VGPR r across each 16-lane half)
    float alpha[8];
#pragma unroll
    for (int r = 0; r < 8; ++r) {
      float mx = s[0][r] * scale;
#pragma unroll
      for (int nt = 0; nt < 4; ++nt) {
        s[nt][r] *= scale;
        mx = fmaxf(mx, s[nt][r]);
      }
#pragma unroll
      for (int off = 8; off >= 1; off >>= 1) mx = fmaxf(mx, __shfl_xor(mx, off, 32));
      float mnew = fmaxf(mrun[r], mx);
      alpha[r] = __expf(mrun[r] - mnew);
      mrun[r] = mnew;
      float sum = 0.f;
#pragma unroll
      for (int nt = 0; nt < 4; ++nt) {
        float p = __expf(s[nt][r] - mnew);
        s[nt][r] = p;
        sum += p;
      }
#pragma unroll
      for (int off = 8; off >= 1; off >>= 1) sum += __shfl_xor(sum, off, 32);
      lrun[r] = lrun[r] * alpha[r] + sum;
    }
#pragma unroll
    for (int dt = 0; dt < 8; ++dt)
#pragma unroll
      for (int r = 0; r < 8; ++r) o[dt][r] *= alpha[r];

    // stage P (C-layout -> LDS row-major) for re-swizzle into A fragments
#pragma unroll
    for (int nt = 0; nt < 4; ++nt)
#pragma unroll
      for (int r = 0; r < 8; ++r)
        ldsP[wid][(rh + r) * 72 + nt * 16 + (lane & 15)] = f2bf(s[nt][r]);

    // O += P(16x64) * V(64x128)
#pragma unroll
    for (int ks = 0; ks < 2; ++ks) {
      v16bf pf = frag_a(&ldsP[wid][(lane & 15) * 72 + ks * 32 + (lane >> 4) * 16]);
#pragma unroll
      for (int dt = 0; dt < 8; ++dt) {
        v16bf vf = frag_b(ldsVt + (dt * 16 + (lane & 15)) * 72 + ks * 32 + (lane >> 4) * 16);
        o[dt] = wmma_bf16(pf, vf, o[dt]);
      }
    }
    __syncthreads();
  }

  // epilogue: normalize and store bf16 into (B,S,D) layout
#pragma unroll
  for (int dt = 0; dt < 8; ++dt) {
#pragma unroll
    for (int r = 0; r < 8; ++r) {
      int row = m0 + rh + r;
      int col = (int)headoff + dt * 16 + (lane & 15);
      O[(size_t)(b * Sq + row) * Dm + col] = f2bf(o[dt][r] / lrun[r]);
    }
  }
}

// ------------------------------------------------------------------- launcher
extern "C" void kernel_launch(void* const* d_in, const int* in_sizes, int n_in,
                              void* d_out, int out_size, void* d_ws, size_t ws_size,
                              hipStream_t stream) {
  const float* x   = (const float*)d_in[0];
  const float* fc  = (const float*)d_in[1];
  const float* fs  = (const float*)d_in[2];
  const float* qw  = (const float*)d_in[3];
  const float* qb  = (const float*)d_in[4];
  const float* kw  = (const float*)d_in[5];
  const float* kb  = (const float*)d_in[6];
  const float* vw  = (const float*)d_in[7];
  const float* vb  = (const float*)d_in[8];
  const float* ow  = (const float*)d_in[9];
  const float* ob  = (const float*)d_in[10];
  const float* qnw = (const float*)d_in[11];
  const float* knw = (const float*)d_in[12];
  float* out = (float*)d_out;

  const size_t M  = (size_t)Bsz * Sq;   // 4096
  const size_t MD = M * Dm;             // 8,388,608
  const size_t WD = (size_t)Dm * Dm;    // 4,194,304

  unsigned short* xb  = (unsigned short*)d_ws;
  unsigned short* qwb = xb + MD;
  unsigned short* kwb = qwb + WD;
  unsigned short* vwb = kwb + WD;
  unsigned short* owb = vwb + WD;
  unsigned short* q16 = owb + WD;
  unsigned short* k16 = q16 + MD;
  unsigned short* v16 = k16 + MD;
  unsigned short* a16 = v16 + MD;
  float* qf32 = (float*)(a16 + MD);
  float* kf32 = qf32 + MD;

  cvt4_kernel<<<(int)(MD / 1024), 256, 0, stream>>>(x, xb, (int)(MD / 4));
  cvt4_kernel<<<(int)(WD / 1024), 256, 0, stream>>>(qw, qwb, (int)(WD / 4));
  cvt4_kernel<<<(int)(WD / 1024), 256, 0, stream>>>(kw, kwb, (int)(WD / 4));
  cvt4_kernel<<<(int)(WD / 1024), 256, 0, stream>>>(vw, vwb, (int)(WD / 4));
  cvt4_kernel<<<(int)(WD / 1024), 256, 0, stream>>>(ow, owb, (int)(WD / 4));

  dim3 gg((unsigned)(M / 128), (unsigned)(Dm / 128));
  gemm_bf16_wmma<<<gg, 256, 0, stream>>>(xb, qwb, qb, qf32, (int)M, Dm, Dm, 0);
  gemm_bf16_wmma<<<gg, 256, 0, stream>>>(xb, kwb, kb, kf32, (int)M, Dm, Dm, 0);
  gemm_bf16_wmma<<<gg, 256, 0, stream>>>(xb, vwb, vb, v16,  (int)M, Dm, Dm, 1);

  rmsnorm_rope_kernel<<<(int)M, 256, 0, stream>>>(qf32, qnw, fc, fs, q16);
  rmsnorm_rope_kernel<<<(int)M, 256, 0, stream>>>(kf32, knw, fc, fs, k16);

  attn_wmma<<<Bsz * Hh * (Sq / 128), 256, 0, stream>>>(q16, k16, v16, a16);

  gemm_bf16_wmma<<<gg, 256, 0, stream>>>(a16, owb, ob, out, (int)M, Dm, Dm, 0);
}
